// GPTModule_12824772346357
// MI455X (gfx1250) — matrix operand order repaired and checked by reading, compiled, not run on hardware
//
#include <hip/hip_runtime.h>
#include <math.h>

// ---- model dims ----
constexpr int NL = 8;
constexpr int NB = 2;
constexpr int NT = 1024;
constexpr int NC = 768;
constexpr int NH = 12;
constexpr int ND = 64;          // NC / NH
constexpr int NV = 32000;
constexpr int NROW = NB * NT;   // 2048
constexpr int NC4 = 4 * NC;     // 3072

typedef _Float16 v16h __attribute__((ext_vector_type(16)));
typedef float    v8f  __attribute__((ext_vector_type(8)));
typedef unsigned int u32x4 __attribute__((ext_vector_type(4)));
typedef int          i32x4 __attribute__((ext_vector_type(4)));
typedef int          i32x8 __attribute__((ext_vector_type(8)));

// TDM builtin exists with different arity per toolchain:
//   ROCm 7.2 / clang-22 : 5 args (g0, g1, i32x4, i32x4, cpol)
//   amdgpu-toolchain / clang-23 : 6 args (g0, g1, i32x4, i32x4, i32x8, cpol)
#if defined(__has_builtin)
#if __has_builtin(__builtin_amdgcn_tensor_load_to_lds)
#define USE_TDM 1
#if __clang_major__ >= 23
#define TDM_6ARG 1
#else
#define TDM_6ARG 0
#endif
#endif
#endif
#ifndef USE_TDM
#define USE_TDM 0
#endif

#if USE_TDM
#if TDM_6ARG
#pragma message("CDNA5: TDM path ENABLED (6-arg builtin, clang>=23)")
#else
#pragma message("CDNA5: TDM path ENABLED (5-arg builtin, clang<23)")
#endif
#else
#pragma message("CDNA5: TDM path DISABLED (builtin not found) - VGPR staging fallback")
#endif

union Frag16 {
    v16h v;
    _Float16 h[16];
    unsigned int u[8];
};

__device__ inline v8f wmma_f16(v16h a, v16h b, v8f c) {
    // D = A(16x32 f16) * B(32x16 f16) + C(16x16 f32)
    return __builtin_amdgcn_wmma_f32_16x16x32_f16(
        /*neg_a=*/false, a, /*neg_b=*/false, b,
        /*c_mod=*/(short)0, c, /*reuse_a=*/false, /*reuse_b=*/false);
}

// ---------------------------------------------------------------------------
// Embedding: x[b,t,:] = tok_emb[idx[b,t],:] + pos_emb[t,:]   (fp32)
// ---------------------------------------------------------------------------
__global__ void embed_kernel(const int* __restrict__ idx,
                             const float* __restrict__ tok,
                             const float* __restrict__ pos,
                             float* __restrict__ x, int total) {
    int g = blockIdx.x * blockDim.x + threadIdx.x;
    if (g >= total) return;
    int c  = g % NC;
    int bt = g / NC;
    int t  = bt % NT;
    x[g] = tok[(size_t)idx[bt] * NC + c] + pos[(size_t)t * NC + c];
}

// ---------------------------------------------------------------------------
// LayerNorm (fp32 in) -> f16 out.  One 128-thread block per row, C = 768.
// ---------------------------------------------------------------------------
__global__ __launch_bounds__(128) void ln_f16_kernel(
    const float* __restrict__ x, const float* __restrict__ w,
    const float* __restrict__ b, _Float16* __restrict__ out, int C) {
    int row = blockIdx.x;
    const float* xr = x + (size_t)row * C;
    float s = 0.f, sq = 0.f;
    for (int c = threadIdx.x; c < C; c += 128) {
        float v = xr[c];
        s += v; sq += v * v;
    }
    for (int o = 16; o > 0; o >>= 1) {
        s  += __shfl_down(s, o);
        sq += __shfl_down(sq, o);
    }
    __shared__ float red[8];
    int wv = threadIdx.x >> 5;
    if ((threadIdx.x & 31) == 0) { red[wv] = s; red[4 + wv] = sq; }
    __syncthreads();
    s  = red[0] + red[1] + red[2] + red[3];
    sq = red[4] + red[5] + red[6] + red[7];
    float mean = s / C;
    float var  = sq / C - mean * mean;
    float rstd = rsqrtf(var + 1e-5f);
    _Float16* orow = out + (size_t)row * C;
    for (int c = threadIdx.x; c < C; c += 128)
        orow[c] = (_Float16)(((xr[c] - mean) * rstd) * w[c] + b[c]);
}

// ---------------------------------------------------------------------------
// WMMA GEMM: out(M,N) = A(M,K; f16) * Bw(K,N; fp32 -> f16 in LDS) + bias
// Block tile 64x64, BK=32; 128 threads = 4 waves; each wave: 2x2 WMMA tiles.
// A tile is staged by the Tensor Data Mover (one descriptor per 64x32 tile,
// HW pad = +1 DWORD per 16-DWORD row -> LDS row stride 34 halves).  B tile is
// staged through VGPRs with fused fp32->f16 conversion.
// EPI: 0 = store f16, 1 = exact GELU -> f16, 2 = residual add -> fp32 (io),
//      3 = plain fp32 store (io).
// ---------------------------------------------------------------------------
template <int EPI>
__global__ __launch_bounds__(128) void gemm_wmma(
    const _Float16* __restrict__ A, const float* __restrict__ Bw,
    const float* __restrict__ bias, float* __restrict__ io,
    _Float16* __restrict__ outH, int M, int N, int K) {
    constexpr int LDA = 34;   // 32 + pad (halves) == TDM pad_interval/amount
    constexpr int LDB = 66;   // 64 + pad (halves)
    __shared__ _Float16 sA[64 * LDA];
    __shared__ _Float16 sB[32 * LDB];

    int tid  = threadIdx.x;
    int wave = tid >> 5, lane = tid & 31;
    int lm = lane & 15, hf = lane >> 4;
    int waveM = wave >> 1, waveN = wave & 1;
    int rowBase = blockIdx.y * 64;
    int colBase = blockIdx.x * 64;

    Frag16 a[2], bf[2];
    v8f c[2][2] = {};

#if USE_TDM
    unsigned ldsAoff = (unsigned)(size_t)&sA[0];
    unsigned long long abase =
        (unsigned long long)(const void*)(A + (size_t)rowBase * K);
#endif

    int nk = K >> 5;
    for (int kt = 0; kt < nk; kt++) {
#if USE_TDM
        // ---- stage A tile (f16, 64 rows x 32 halves) via TDM
        if (wave == 0) {
            unsigned long long ga = abase + (unsigned long long)kt * 64ull;
            u32x4 g0;
            g0[0] = 1u;                                   // count=1 (valid D#)
            g0[1] = ldsAoff;                              // lds_addr (bytes)
            g0[2] = (unsigned)(ga & 0xffffffffull);       // global_addr lo
            g0[3] = (unsigned)((ga >> 32) & 0x1ffffffull) // global_addr hi
                    | (2u << 30);                         // type = 2 (image)
            i32x8 g1;
            g1[0] = (int)((1u << 16)    // data_size = 2 bytes
                        | (1u << 20)    // pad_enable
                        | (3u << 22));  // pad_interval: 16 DWORDs (64B row)
                                        // pad_amount field 0 -> 1 DWORD pad
            g1[1] = (int)(((unsigned)K & 0xffffu) << 16);           // dim0 lo
            g1[2] = (int)(((unsigned)K >> 16) |
                          (((unsigned)M & 0xffffu) << 16));         // dim0 hi, dim1 lo
            g1[3] = (int)(((unsigned)M >> 16) | (32u << 16));       // dim1 hi, tile_dim0=32
            g1[4] = 64;                                             // tile_dim1=64
            g1[5] = K;                                              // dim0_stride lo
            g1[6] = 0;                                              // dim0_stride hi
            g1[7] = 0;
            i32x4 z4 = {0, 0, 0, 0};
#if TDM_6ARG
            i32x8 z8 = {0, 0, 0, 0, 0, 0, 0, 0};
            __builtin_amdgcn_tensor_load_to_lds(g0, g1, z4, z4, z8, 0);
#else
            __builtin_amdgcn_tensor_load_to_lds(g0, g1, z4, z4, 0);
#endif
            __builtin_amdgcn_s_wait_tensorcnt(0);
        }
#else
        // ---- stage A tile (f16, 64x32): each thread 16 contiguous halves
        {
            int r = tid >> 1, kk = (tid & 1) << 4;
            const _Float16* src = A + (size_t)(rowBase + r) * K + kt * 32 + kk;
            uint4 d0 = ((const uint4*)src)[0];
            uint4 d1 = ((const uint4*)src)[1];
            unsigned int* dst = (unsigned int*)&sA[r * LDA + kk];
            dst[0] = d0.x; dst[1] = d0.y; dst[2] = d0.z; dst[3] = d0.w;
            dst[4] = d1.x; dst[5] = d1.y; dst[6] = d1.z; dst[7] = d1.w;
        }
#endif
        // ---- stage B tile (fp32 -> f16, 32x64): each thread 16 floats
        {
            int r = tid >> 2, cc = (tid & 3) << 4;
            const float* src = Bw + (size_t)(kt * 32 + r) * N + colBase + cc;
            if (kt + 1 < nk)
                __builtin_prefetch(src + (size_t)32 * N, 0, 1);  // next k-tile
            _Float16* dst = &sB[r * LDB + cc];
#pragma unroll
            for (int j = 0; j < 16; j += 4) {
                float4 f = *(const float4*)(src + j);
                dst[j]     = (_Float16)f.x;
                dst[j + 1] = (_Float16)f.y;
                dst[j + 2] = (_Float16)f.z;
                dst[j + 3] = (_Float16)f.w;
            }
        }
        __syncthreads();

        // ---- A fragments (lane = row; K = 16h + 0..15)
#pragma unroll
        for (int rt = 0; rt < 2; rt++) {
            const _Float16* ar = &sA[(waveM * 32 + rt * 16 + lm) * LDA];
#pragma unroll
            for (int v = 0; v < 4; v++)
                a[rt].u[v] = *(const unsigned int*)(ar + 16 * hf + 2 * v);
#pragma unroll
            for (int v = 0; v < 4; v++)
                a[rt].u[4 + v] = *(const unsigned int*)(ar + 16 * hf + 8 + 2 * v);
        }
        // ---- B fragments (lane = col; K = 16h + 0..15)
#pragma unroll
        for (int ct = 0; ct < 2; ct++) {
            int colL = waveN * 32 + ct * 16 + lm;
#pragma unroll
            for (int v = 0; v < 8; v++) {
                int k0 = 16 * hf + 2 * v;
                bf[ct].h[2 * v]     = sB[k0 * LDB + colL];
                bf[ct].h[2 * v + 1] = sB[(k0 + 1) * LDB + colL];
            }
        }
        c[0][0] = wmma_f16(a[0].v, bf[0].v, c[0][0]);
        c[0][1] = wmma_f16(a[0].v, bf[1].v, c[0][1]);
        c[1][0] = wmma_f16(a[1].v, bf[0].v, c[1][0]);
        c[1][1] = wmma_f16(a[1].v, bf[1].v, c[1][1]);
        __syncthreads();
    }

    // ---- epilogue (D layout: VGPR i -> row i + 8*hf, col = lane&15)
#pragma unroll
    for (int rt = 0; rt < 2; rt++)
#pragma unroll
        for (int ct = 0; ct < 2; ct++)
#pragma unroll
            for (int i = 0; i < 8; i++) {
                int row = rowBase + waveM * 32 + rt * 16 + i + 8 * hf;
                int col = colBase + waveN * 32 + ct * 16 + lm;
                float vsum = c[rt][ct][i];
                if (bias) vsum += bias[col];
                size_t o = (size_t)row * N + col;
                if (EPI == 0) {
                    outH[o] = (_Float16)vsum;
                } else if (EPI == 1) {
                    float g = 0.5f * vsum * (1.0f + erff(vsum * 0.70710678118654752f));
                    outH[o] = (_Float16)g;
                } else if (EPI == 2) {
                    io[o] = io[o] + vsum;
                } else {
                    io[o] = vsum;
                }
            }
}

// ---------------------------------------------------------------------------
// Flash-style causal attention.  q,k,v,y : f16 [B, T, C] with head layout
// (h*64 + d).  Grid (T/64, H, B), 128 threads; each wave owns a 16-query tile
// and streams 32-key blocks with online softmax.  All matmuls via WMMA f16.
// ---------------------------------------------------------------------------
__global__ __launch_bounds__(128) void attn_kernel(
    const _Float16* __restrict__ Q, const _Float16* __restrict__ K,
    const _Float16* __restrict__ V, _Float16* __restrict__ Y) {
    __shared__ _Float16 ldsP[4 * 16 * 32];   // per-wave 16x32 P tile

    int wave = threadIdx.x >> 5, lane = threadIdx.x & 31;
    int lm = lane & 15, hf = lane >> 4;
    int qt = blockIdx.x * 4 + wave;          // query tile (16 rows)
    int qbase = qt * 16;
    int b = blockIdx.z, hd = blockIdx.y;
    size_t baseBT = (size_t)b * NT * NC + (size_t)hd * ND;
    _Float16* pWave = &ldsP[wave * 16 * 32];

    // ---- Q fragments, 2 K-chunks of 32 over D=64
    Frag16 qa[2];
    {
        const _Float16* qp = Q + baseBT + (size_t)(qbase + lm) * NC;
#pragma unroll
        for (int ck = 0; ck < 2; ck++) {
#pragma unroll
            for (int v = 0; v < 4; v++)
                qa[ck].u[v] = *(const unsigned int*)(qp + 32 * ck + 16 * hf + 2 * v);
#pragma unroll
            for (int v = 0; v < 4; v++)
                qa[ck].u[4 + v] = *(const unsigned int*)(qp + 32 * ck + 16 * hf + 8 + 2 * v);
        }
    }

    v8f acc[4] = {};
    float mrun[8], lrun[8];
#pragma unroll
    for (int i = 0; i < 8; i++) { mrun[i] = -1e30f; lrun[i] = 0.f; }
    const float scale = 0.125f;              // 1/sqrt(64)

    int nkb = qt / 2 + 1;                    // causal: key blocks of 32
    for (int kb = 0; kb < nkb; kb++) {
        int kb32 = kb * 32;
        // ---- K^T fragments: B[d, key]; lane = key col, 16 contiguous d-vals
        Frag16 kf[2][2];
#pragma unroll
        for (int ct = 0; ct < 2; ct++) {
            const _Float16* kp = K + baseBT + (size_t)(kb32 + ct * 16 + lm) * NC;
#pragma unroll
            for (int ck = 0; ck < 2; ck++) {
                const uint4* s = (const uint4*)(kp + ck * 32 + hf * 16);
                uint4 t0 = s[0], t1 = s[1];
                kf[ct][ck].u[0] = t0.x; kf[ct][ck].u[1] = t0.y;
                kf[ct][ck].u[2] = t0.z; kf[ct][ck].u[3] = t0.w;
                kf[ct][ck].u[4] = t1.x; kf[ct][ck].u[5] = t1.y;
                kf[ct][ck].u[6] = t1.z; kf[ct][ck].u[7] = t1.w;
            }
        }
        v8f s0 = {}, s1 = {};
        s0 = wmma_f16(qa[0].v, kf[0][0].v, s0);
        s0 = wmma_f16(qa[1].v, kf[0][1].v, s0);
        s1 = wmma_f16(qa[0].v, kf[1][0].v, s1);
        s1 = wmma_f16(qa[1].v, kf[1][1].v, s1);

        // ---- mask + scale + row max (wave32 halves reduce independently)
        float rm[8];
#pragma unroll
        for (int i = 0; i < 8; i++) {
            int qrow = qbase + i + 8 * hf;
            int k0 = kb32 + lm, k1 = kb32 + 16 + lm;
            float a0 = (k0 <= qrow) ? s0[i] * scale : -1e30f;
            float a1 = (k1 <= qrow) ? s1[i] * scale : -1e30f;
            s0[i] = a0; s1[i] = a1;
            float m = fmaxf(a0, a1);
            m = fmaxf(m, __shfl_xor(m, 1));
            m = fmaxf(m, __shfl_xor(m, 2));
            m = fmaxf(m, __shfl_xor(m, 4));
            m = fmaxf(m, __shfl_xor(m, 8));
            rm[i] = m;
        }
        float corr[8];
#pragma unroll
        for (int i = 0; i < 8; i++) {
            float mn = fmaxf(mrun[i], rm[i]);
            corr[i] = expf(mrun[i] - mn);
            mrun[i] = mn;
        }
        // ---- p = exp(s - m), row sums, rescale running state
#pragma unroll
        for (int i = 0; i < 8; i++) {
            float p0 = expf(s0[i] - mrun[i]);
            float p1 = expf(s1[i] - mrun[i]);
            s0[i] = p0; s1[i] = p1;
            float r = p0 + p1;
            r += __shfl_xor(r, 1);
            r += __shfl_xor(r, 2);
            r += __shfl_xor(r, 4);
            r += __shfl_xor(r, 8);
            lrun[i] = lrun[i] * corr[i] + r;
            acc[0][i] *= corr[i]; acc[1][i] *= corr[i];
            acc[2][i] *= corr[i]; acc[3][i] *= corr[i];
        }
        // ---- P (D-layout) -> LDS -> A-fragment layout
#pragma unroll
        for (int i = 0; i < 8; i++) {
            int pr = i + 8 * hf;
            pWave[pr * 32 + lm]      = (_Float16)s0[i];
            pWave[pr * 32 + 16 + lm] = (_Float16)s1[i];
        }
        asm volatile("s_wait_dscnt 0" ::: "memory");   // same-wave DS RAW
        Frag16 pf;
        {
            const _Float16* pp = &pWave[lm * 32];
#pragma unroll
            for (int v = 0; v < 4; v++)
                pf.u[v] = *(const unsigned int*)(pp + 16 * hf + 2 * v);
#pragma unroll
            for (int v = 0; v < 4; v++)
                pf.u[4 + v] = *(const unsigned int*)(pp + 16 * hf + 8 + 2 * v);
        }
        asm volatile("" ::: "memory");
        // ---- acc += P(16x32) @ V(32x64): lane = d col, K = key indices
#pragma unroll
        for (int nt = 0; nt < 4; nt++) {
            Frag16 vf;
            const _Float16* vp = V + baseBT + nt * 16 + lm;
#pragma unroll
            for (int v = 0; v < 8; v++) {
                int key0 = kb32 + hf * 16 + 2 * v;
                vf.h[2 * v]     = vp[(size_t)key0 * NC];
                vf.h[2 * v + 1] = vp[(size_t)(key0 + 1) * NC];
            }
            acc[nt] = wmma_f16(pf.v, vf.v, acc[nt]);
        }
    }

    // ---- normalize and store y
#pragma unroll
    for (int nt = 0; nt < 4; nt++)
#pragma unroll
        for (int i = 0; i < 8; i++) {
            int row = qbase + i + 8 * hf;
            Y[baseBT + (size_t)row * NC + nt * 16 + lm] =
                (_Float16)(acc[nt][i] * (1.0f / lrun[i]));
        }
}

// ---------------------------------------------------------------------------
// Host launcher
// ---------------------------------------------------------------------------
extern "C" void kernel_launch(void* const* d_in, const int* in_sizes, int n_in,
                              void* d_out, int out_size, void* d_ws, size_t ws_size,
                              hipStream_t stream) {
    (void)in_sizes; (void)n_in; (void)out_size; (void)ws_size;
    const int*   idx   = (const int*)  d_in[0];
    const float* tok   = (const float*)d_in[1];
    const float* pos   = (const float*)d_in[2];
    const float* Wq    = (const float*)d_in[3];
    const float* Wk    = (const float*)d_in[4];
    const float* Wv    = (const float*)d_in[5];
    const float* Wo    = (const float*)d_in[6];
    const float* bq    = (const float*)d_in[7];
    const float* bk    = (const float*)d_in[8];
    const float* bv    = (const float*)d_in[9];
    const float* bo    = (const float*)d_in[10];
    const float* W1    = (const float*)d_in[11];
    const float* b1    = (const float*)d_in[12];
    const float* W2    = (const float*)d_in[13];
    const float* b2    = (const float*)d_in[14];
    const float* ln1w  = (const float*)d_in[15];
    const float* ln1b  = (const float*)d_in[16];
    const float* ln2w  = (const float*)d_in[17];
    const float* ln2b  = (const float*)d_in[18];
    const float* lnfw  = (const float*)d_in[19];
    const float* lnfb  = (const float*)d_in[20];
    const float* headw = (const float*)d_in[21];

    // workspace carve (all offsets 256B-aligned)
    char* p = (char*)d_ws;
    float*    x   = (float*)p;    p += (size_t)NROW * NC * sizeof(float);
    _Float16* h16 = (_Float16*)p; p += (size_t)NROW * NC * sizeof(_Float16);
    _Float16* q16 = (_Float16*)p; p += (size_t)NROW * NC * sizeof(_Float16);
    _Float16* k16 = (_Float16*)p; p += (size_t)NROW * NC * sizeof(_Float16);
    _Float16* v16 = (_Float16*)p; p += (size_t)NROW * NC * sizeof(_Float16);
    _Float16* y16 = (_Float16*)p; p += (size_t)NROW * NC * sizeof(_Float16);
    _Float16* a16 = (_Float16*)p; p += (size_t)NROW * NC4 * sizeof(_Float16);

    dim3 blk(128);
    dim3 gCC(NC / 64, NROW / 64);

    embed_kernel<<<(NROW * NC + 255) / 256, 256, 0, stream>>>(idx, tok, pos, x, NROW * NC);

    for (int l = 0; l < NL; l++) {
        const size_t wCC = (size_t)l * NC * NC;
        ln_f16_kernel<<<NROW, 128, 0, stream>>>(x, ln1w + l * NC, ln1b + l * NC, h16, NC);
        gemm_wmma<0><<<gCC, blk, 0, stream>>>(h16, Wq + wCC, bq + l * NC, nullptr, q16, NROW, NC, NC);
        gemm_wmma<0><<<gCC, blk, 0, stream>>>(h16, Wk + wCC, bk + l * NC, nullptr, k16, NROW, NC, NC);
        gemm_wmma<0><<<gCC, blk, 0, stream>>>(h16, Wv + wCC, bv + l * NC, nullptr, v16, NROW, NC, NC);
        attn_kernel<<<dim3(NT / 64, NH, NB), blk, 0, stream>>>(q16, k16, v16, y16);
        gemm_wmma<2><<<gCC, blk, 0, stream>>>(y16, Wo + wCC, bo + l * NC, x, nullptr, NROW, NC, NC);
        ln_f16_kernel<<<NROW, 128, 0, stream>>>(x, ln2w + l * NC, ln2b + l * NC, h16, NC);
        gemm_wmma<1><<<dim3(NC4 / 64, NROW / 64), blk, 0, stream>>>(
            h16, W1 + (size_t)l * NC * NC4, b1 + (size_t)l * NC4, nullptr, a16, NROW, NC4, NC);
        gemm_wmma<2><<<gCC, blk, 0, stream>>>(
            a16, W2 + (size_t)l * NC4 * NC, b2 + l * NC, x, nullptr, NROW, NC, NC4);
    }

    ln_f16_kernel<<<NROW, 128, 0, stream>>>(x, lnfw, lnfb, h16, NC);
    gemm_wmma<3><<<dim3(NV / 64, NROW / 64), blk, 0, stream>>>(
        h16, headw, nullptr, (float*)d_out, nullptr, NROW, NV, NC);
}
